// StartPointAttnModel_90074054131989
// MI455X (gfx1250) — compile-verified, matrix operand order checked by reading, not run on hardware
//
#include <hip/hip_runtime.h>
#include <hip/hip_bf16.h>

// ---------------------------------------------------------------------------
// Problem constants (from reference)
// ---------------------------------------------------------------------------
#define T_STEPS   150
#define BATCH     512
#define D_IN      1024
#define D_H       256
#define G4        1024      // 4*D_H
#define VOCAB     3
#define MAXLEN    64
#define TB        (T_STEPS * BATCH)   // 76800

typedef __attribute__((ext_vector_type(16))) __bf16 v16bf;
typedef __attribute__((ext_vector_type(8)))  float  v8f;

union FragAB { uint4 q[2]; unsigned u[8]; v16bf v; };
union Pack16 { __bf16 h[16]; uint4 q[2]; };

// LDS row stride in bf16 elements: 40 (= 20 dwords, 80 B).
//  - every 16B sub-row (offsets 0/16/32/48 B) stays 16B-aligned
//  - b128 access by 16 lanes: starts 20*r mod 64 are 16 distinct multiples
//    of 4 -> all 64 banks covered, conflict-free
#define LDST 40

// ---------------------------------------------------------------------------
// WMMA bf16 GEMM:  out = alpha * (A[M,K] * W[N,K]^T + bias[N]) + Cin[M,N]
//   A: bf16 (Abf, lda) OR fp32 converted on the fly (Af32)
//   out: fp32 (outF) and/or bf16 (outB), leading dim ldc
// Tile: 128x128 per workgroup, 256 threads = 8 waves, wave tile 64x32,
// micro tile 16x16x32 via v_wmma_f32_16x16x32_bf16. K multiple of 32;
// M,N multiples of 128 (as launched).
//
// Double buffering uses GLOBAL_LOAD_ASYNC_TO_LDS_B128 (ASYNCcnt-tracked DMA,
// no VGPR staging -> no spills): issue async loads of the next K-chunk into
// lds[buf^1] before the WMMA block, s_wait_asynccnt 0 after it.
// ---------------------------------------------------------------------------
__global__ __launch_bounds__(256, 1)
void gemm_wmma_bf16(const __bf16* __restrict__ Abf, const float* __restrict__ Af32, long lda,
                    const __bf16* __restrict__ W, long ldw,
                    const float* __restrict__ bias,
                    const __bf16* __restrict__ Cin, long ldcin,
                    float alpha,
                    float* __restrict__ outF, __bf16* __restrict__ outB, long ldc,
                    int M, int N, int K)
{
    __shared__ __bf16 ldsA[2][128][LDST];
    __shared__ __bf16 ldsB[2][128][LDST];

    const int tid   = threadIdx.x;
    const int wid   = tid >> 5;
    const int lane  = tid & 31;
    const int half  = lane >> 4;      // 0: K 0..7/16..23, 1: K 8..15/24..31
    const int lm    = lane & 15;
    const int waveM = (wid >> 2) * 64;    // 0 or 64
    const int waveN = (wid & 3)  * 32;    // 0,32,64,96
    const long rowBase = (long)blockIdx.y * 128;
    const long colBase = (long)blockIdx.x * 128;

    v8f acc[4][2];
#pragma unroll
    for (int mt = 0; mt < 4; mt++)
#pragma unroll
        for (int nt = 0; nt < 2; nt++)
#pragma unroll
            for (int r = 0; r < 8; r++) acc[mt][nt][r] = 0.0f;

    // cooperative tile staging: each thread owns 16 contiguous bf16 of a row
    const int ldRow = tid >> 1;           // 0..127
    const int ldCol = (tid & 1) * 16;     // 0 or 16

    // ---- async DMA staging: global -> LDS, no VGPR round-trip ----
    // INST_OFFSET applies to both LDS and memory address (ISA 8.4.4), so each
    // thread's 32B slice is two b128 ops sharing the same address operands.
    auto asyncA = [&](int nbuf, int kc) {
        const unsigned laddr = (unsigned)(size_t)&ldsA[nbuf][ldRow][ldCol];
        const unsigned goff  = (unsigned)(((rowBase + ldRow) * lda + kc + ldCol) * 2);
        asm volatile("global_load_async_to_lds_b128 %0, %1, %2\n\t"
                     "global_load_async_to_lds_b128 %0, %1, %2 offset:16"
                     :: "v"(laddr), "v"(goff), "s"(Abf) : "memory");
    };
    auto asyncB = [&](int nbuf, int kc) {
        const unsigned laddr = (unsigned)(size_t)&ldsB[nbuf][ldRow][ldCol];
        const unsigned goff  = (unsigned)(((colBase + ldRow) * ldw + kc + ldCol) * 2);
        asm volatile("global_load_async_to_lds_b128 %0, %1, %2\n\t"
                     "global_load_async_to_lds_b128 %0, %1, %2 offset:16"
                     :: "v"(laddr), "v"(goff), "s"(W) : "memory");
    };
    // fp32 A path (Xproj mega-GEMM only): synchronous load-convert-store;
    // that launch has 4800 blocks, inter-block occupancy hides the latency.
    auto stageA_f32 = [&](int nbuf, int kc) {
        const float* s = Af32 + (rowBase + ldRow) * lda + kc + ldCol;
        const float4 f0 = *(const float4*)(s);
        const float4 f1 = *(const float4*)(s + 4);
        const float4 f2 = *(const float4*)(s + 8);
        const float4 f3 = *(const float4*)(s + 12);
        Pack16 r;
        r.h[0]  = (__bf16)f0.x; r.h[1]  = (__bf16)f0.y; r.h[2]  = (__bf16)f0.z; r.h[3]  = (__bf16)f0.w;
        r.h[4]  = (__bf16)f1.x; r.h[5]  = (__bf16)f1.y; r.h[6]  = (__bf16)f1.z; r.h[7]  = (__bf16)f1.w;
        r.h[8]  = (__bf16)f2.x; r.h[9]  = (__bf16)f2.y; r.h[10] = (__bf16)f2.z; r.h[11] = (__bf16)f2.w;
        r.h[12] = (__bf16)f3.x; r.h[13] = (__bf16)f3.y; r.h[14] = (__bf16)f3.z; r.h[15] = (__bf16)f3.w;
        *(uint4*)&ldsA[nbuf][ldRow][ldCol]     = r.q[0];
        *(uint4*)&ldsA[nbuf][ldRow][ldCol + 8] = r.q[1];
    };

    // ---- prologue: stage first K-chunk into buffer 0 ----
    if (Af32) stageA_f32(0, 0); else asyncA(0, 0);
    asyncB(0, 0);
    asm volatile("s_wait_asynccnt 0" ::: "memory");
    __syncthreads();

    int buf = 0;
    for (int kc = 0; kc < K; kc += 32) {
        const bool hasNext = (kc + 32) < K;
        if (hasNext) {           // kick off next tile's DMA before the math
            if (Af32) stageA_f32(buf ^ 1, kc + 32); else asyncA(buf ^ 1, kc + 32);
            asyncB(buf ^ 1, kc + 32);
        }

        // ---- gather fragments (two ds_load_b128 each) + 8 WMMA ----
        FragAB a[4], b[2];
#pragma unroll
        for (int nt = 0; nt < 2; nt++) {
            const int row = waveN + nt * 16 + lm;
            b[nt].q[0] = *(const uint4*)&ldsB[buf][row][half * 8];
            b[nt].q[1] = *(const uint4*)&ldsB[buf][row][16 + half * 8];
        }
#pragma unroll
        for (int mt = 0; mt < 4; mt++) {
            const int row = waveM + mt * 16 + lm;
            a[mt].q[0] = *(const uint4*)&ldsA[buf][row][half * 8];
            a[mt].q[1] = *(const uint4*)&ldsA[buf][row][16 + half * 8];
        }
#pragma unroll
        for (int mt = 0; mt < 4; mt++)
#pragma unroll
            for (int nt = 0; nt < 2; nt++)
                acc[mt][nt] = __builtin_amdgcn_wmma_f32_16x16x32_bf16(
                    false, a[mt].v, false, b[nt].v, (short)0, acc[mt][nt], false, false);

        if (hasNext)             // DMA completion wait lands after the WMMAs
            asm volatile("s_wait_asynccnt 0" ::: "memory");
        __syncthreads();
        buf ^= 1;
    }

    // ---- epilogue: D layout = lane n (0..15), vgpr r -> M = half*8 + r ----
#pragma unroll
    for (int mt = 0; mt < 4; mt++) {
#pragma unroll
        for (int nt = 0; nt < 2; nt++) {
            const long gn = colBase + waveN + nt * 16 + lm;
            const float bv = bias ? bias[gn] : 0.0f;
#pragma unroll
            for (int r = 0; r < 8; r++) {
                const long gm = rowBase + waveM + mt * 16 + half * 8 + r;
                if (gm < M && gn < N) {
                    float v = (acc[mt][nt][r] + bv) * alpha;
                    if (Cin)  v += (float)Cin[gm * ldcin + gn];
                    if (outF) outF[gm * ldc + gn] = v;
                    if (outB) outB[gm * ldc + gn] = (__bf16)v;
                }
            }
        }
    }
}

// ---------------------------------------------------------------------------
// LSTM pointwise cell: gates[B,4H] (i,f,g,o packing), updates c, emits h.
// Optionally emits relu(h) as fp32 + bf16 (decoder).
// ---------------------------------------------------------------------------
__device__ __forceinline__ float sigf(float x) { return 1.0f / (1.0f + __expf(-x)); }

__global__ __launch_bounds__(256)
void lstm_cell_kernel(const float* __restrict__ gates, float* __restrict__ c,
                      __bf16* __restrict__ h_out, long ldh,
                      float* __restrict__ relu_f, __bf16* __restrict__ relu_b, long ldo)
{
    const int b = blockIdx.x;
    const int d = threadIdx.x;
    const long gb = (long)b * G4;
    const float gi = gates[gb + d];
    const float gf = gates[gb + D_H + d];
    const float gg = gates[gb + 2 * D_H + d];
    const float go = gates[gb + 3 * D_H + d];
    const long idx = (long)b * D_H + d;
    float cv = sigf(gf) * c[idx] + sigf(gi) * tanhf(gg);
    c[idx] = cv;
    const float h = sigf(go) * tanhf(cv);
    h_out[(long)b * ldh + d] = (__bf16)h;
    if (relu_f) {
        const float r = fmaxf(h, 0.0f);
        relu_f[idx] = r;
        relu_b[(long)b * ldo + d] = (__bf16)r;
    }
}

// ---------------------------------------------------------------------------
// Fused attention per batch element: scores over T, softmax, weighted V sum.
// k/v are bf16 [T,B,D_H] (both L2-resident: 2 x 39 MB < 192 MB L2).
// grid = BATCH, block = 256.
// ---------------------------------------------------------------------------
__global__ __launch_bounds__(256)
void attn_kernel(const float* __restrict__ q, const __bf16* __restrict__ kp,
                 const __bf16* __restrict__ vp, __bf16* __restrict__ ctx)
{
    __shared__ float qs[256], sc[256], red[256];
    const int b = blockIdx.x;
    const int tid = threadIdx.x;
    qs[tid] = q[(long)b * D_H + tid];
    __syncthreads();

    float s = -1e30f;
    if (tid < T_STEPS) {
        const __bf16* krow = kp + ((long)tid * BATCH + b) * D_H;
        float acc = 0.0f;
#pragma unroll 8
        for (int d = 0; d < D_H; d++) acc += (float)krow[d] * qs[d];
        s = acc;
    }
    sc[tid] = s;
    red[tid] = s;
    for (int st = 128; st > 0; st >>= 1) {
        __syncthreads();
        if (tid < st) red[tid] = fmaxf(red[tid], red[tid + st]);
    }
    __syncthreads();
    const float m = red[0];
    const float e = (tid < T_STEPS) ? __expf(sc[tid] - m) : 0.0f;
    __syncthreads();
    red[tid] = e;
    sc[tid]  = e;
    for (int st = 128; st > 0; st >>= 1) {
        __syncthreads();
        if (tid < st) red[tid] += red[tid + st];
    }
    __syncthreads();
    const float inv = 1.0f / red[0];

    // ctx[d] = sum_t p[t] * v[t,b,d]  -- consecutive lanes read consecutive d
    float acc = 0.0f;
    for (int t = 0; t < T_STEPS; t++)
        acc += sc[t] * (float)vp[((long)t * BATCH + b) * D_H + tid];
    ctx[(long)b * D_H + tid] = (__bf16)(acc * inv);
}

// ---------------------------------------------------------------------------
// Tiny output head: logits[b,v] = out[b,:] . lin_w[v,:] + lin_b[v]
// ---------------------------------------------------------------------------
__global__ void logits_kernel(const float* __restrict__ out, const float* __restrict__ lin_w,
                              const float* __restrict__ lin_b, float* __restrict__ dst)
{
    const int idx = blockIdx.x * blockDim.x + threadIdx.x;
    if (idx >= BATCH * VOCAB) return;
    const int b = idx / VOCAB, v = idx % VOCAB;
    float s = lin_b[v];
#pragma unroll 8
    for (int d = 0; d < D_H; d++) s += out[(long)b * D_H + d] * lin_w[v * D_H + d];
    dst[idx] = s;
}

// ---------------------------------------------------------------------------
// Prep kernels
// ---------------------------------------------------------------------------
__global__ void cvt_f32_bf16(const float* __restrict__ src, __bf16* __restrict__ dst, long n)
{
    const long i = (long)blockIdx.x * blockDim.x + threadIdx.x;
    if (i < n) dst[i] = (__bf16)src[i];
}

__global__ void add_vec(const float* __restrict__ a, const float* __restrict__ b,
                        float* __restrict__ dst, int n)
{
    const int i = blockIdx.x * blockDim.x + threadIdx.x;
    if (i < n) dst[i] = a[i] + b[i];
}

// combined decoder weight [dec_Wih | dec_Whh] -> (1024 x 768) bf16
__global__ void build_decw(const float* __restrict__ wih, const float* __restrict__ whh,
                           __bf16* __restrict__ dst)
{
    const long i = (long)blockIdx.x * blockDim.x + threadIdx.x;
    if (i >= (long)G4 * 768) return;
    const int n = (int)(i / 768), j = (int)(i % 768);
    const float v = (j < 512) ? wih[(long)n * 512 + j] : whh[(long)n * D_H + (j - 512)];
    dst[i] = (__bf16)v;
}

__global__ void zero_f32(float* __restrict__ p, long n)
{
    const long i = (long)blockIdx.x * blockDim.x + threadIdx.x;
    if (i < n) p[i] = 0.0f;
}

__global__ void zero_bf16(__bf16* __restrict__ p, long n)
{
    const long i = (long)blockIdx.x * blockDim.x + threadIdx.x;
    if (i < n) p[i] = (__bf16)0.0f;
}

// decoder init: xin[:,0:256] = 0 (out0), xin[:,512:768] = enc h_{T-1}
__global__ __launch_bounds__(256)
void init_decode(__bf16* __restrict__ xin, const __bf16* __restrict__ h_last)
{
    const int b = blockIdx.x, d = threadIdx.x;
    xin[(long)b * 768 + d] = (__bf16)0.0f;
    xin[(long)b * 768 + 512 + d] = h_last[(long)b * D_H + d];
}

// ---------------------------------------------------------------------------
// Host orchestration
// ---------------------------------------------------------------------------
extern "C" void kernel_launch(void* const* d_in, const int* in_sizes, int n_in,
                              void* d_out, int out_size, void* d_ws, size_t ws_size,
                              hipStream_t stream)
{
    const float* x        = (const float*)d_in[0];
    const float* enc_Wih  = (const float*)d_in[1];
    const float* enc_Whh  = (const float*)d_in[2];
    const float* enc_bih  = (const float*)d_in[3];
    const float* enc_bhh  = (const float*)d_in[4];
    const float* dec_Wih  = (const float*)d_in[5];
    const float* dec_Whh  = (const float*)d_in[6];
    const float* dec_bih  = (const float*)d_in[7];
    const float* dec_bhh  = (const float*)d_in[8];
    const float* attn_in_w  = (const float*)d_in[9];
    const float* attn_in_b  = (const float*)d_in[10];
    const float* attn_out_w = (const float*)d_in[11];
    const float* attn_out_b = (const float*)d_in[12];
    const float* lin_w    = (const float*)d_in[13];
    const float* lin_b    = (const float*)d_in[14];
    float* logits_out = (float*)d_out;

    // ---- carve workspace ----
    char* p = (char*)d_ws;
    auto carve = [&](size_t bytes) -> void* {
        void* r = (void*)p;
        p += (bytes + 255) & ~(size_t)255;
        return r;
    };
    __bf16* Xproj   = (__bf16*)carve((size_t)TB * G4 * 2);        // 150 MB
    __bf16* enc     = (__bf16*)carve((size_t)TB * D_H * 2);       // 39 MB (encoder hidden states)
    __bf16* kbf     = (__bf16*)carve((size_t)TB * D_H * 2);       // 39 MB
    __bf16* vbf     = (__bf16*)carve((size_t)TB * D_H * 2);       // 39 MB
    float*  gates   = (float*) carve((size_t)BATCH * G4 * 4);
    float*  cbuf    = (float*) carve((size_t)BATCH * D_H * 4);
    float*  qbuf    = (float*) carve((size_t)BATCH * D_H * 4);
    float*  outf    = (float*) carve((size_t)BATCH * D_H * 4);
    __bf16* ctxb    = (__bf16*)carve((size_t)BATCH * D_H * 2);
    __bf16* xin     = (__bf16*)carve((size_t)BATCH * 768 * 2);    // [out | ctx | h]
    __bf16* hzero   = (__bf16*)carve((size_t)BATCH * D_H * 2);
    __bf16* wih_b   = (__bf16*)carve((size_t)G4 * D_IN * 2);
    __bf16* whh_b   = (__bf16*)carve((size_t)G4 * D_H * 2);
    __bf16* decw_b  = (__bf16*)carve((size_t)G4 * 768 * 2);
    __bf16* attnw_b = (__bf16*)carve((size_t)768 * D_H * 2);
    __bf16* attow_b = (__bf16*)carve((size_t)D_H * D_H * 2);
    float*  encb    = (float*) carve((size_t)G4 * 4);
    float*  decb    = (float*) carve((size_t)G4 * 4);
    (void)ws_size; (void)in_sizes; (void)n_in; (void)out_size;

    const dim3 blk(256);
    auto nb = [](long n) { return dim3((unsigned)((n + 255) / 256)); };

    // ---- prep: weight conversion, bias fusion, state init ----
    cvt_f32_bf16<<<nb((long)G4 * D_IN), blk, 0, stream>>>(enc_Wih, wih_b, (long)G4 * D_IN);
    cvt_f32_bf16<<<nb((long)G4 * D_H),  blk, 0, stream>>>(enc_Whh, whh_b, (long)G4 * D_H);
    cvt_f32_bf16<<<nb((long)768 * D_H), blk, 0, stream>>>(attn_in_w, attnw_b, (long)768 * D_H);
    cvt_f32_bf16<<<nb((long)D_H * D_H), blk, 0, stream>>>(attn_out_w, attow_b, (long)D_H * D_H);
    build_decw<<<nb((long)G4 * 768), blk, 0, stream>>>(dec_Wih, dec_Whh, decw_b);
    add_vec<<<nb(G4), blk, 0, stream>>>(enc_bih, enc_bhh, encb, G4);
    add_vec<<<nb(G4), blk, 0, stream>>>(dec_bih, dec_bhh, decb, G4);
    zero_f32<<<nb((long)BATCH * D_H), blk, 0, stream>>>(cbuf, (long)BATCH * D_H);
    zero_bf16<<<nb((long)BATCH * D_H), blk, 0, stream>>>(hzero, (long)BATCH * D_H);

    // ---- Phase 1: Xproj = x @ enc_Wih^T + (bih+bhh)  (76800x1024x1024) ----
    gemm_wmma_bf16<<<dim3(G4 / 128, TB / 128), blk, 0, stream>>>(
        nullptr, x, D_IN, wih_b, D_IN, encb, nullptr, 0, 1.0f,
        nullptr, Xproj, G4, TB, G4, D_IN);

    // ---- Phase 2: encoder recurrence (serial over T) ----
    for (int t = 0; t < T_STEPS; t++) {
        const __bf16* hprev = (t == 0) ? hzero : (enc + (size_t)(t - 1) * BATCH * D_H);
        gemm_wmma_bf16<<<dim3(G4 / 128, BATCH / 128), blk, 0, stream>>>(
            hprev, nullptr, D_H, whh_b, D_H, nullptr,
            Xproj + (size_t)t * BATCH * G4, G4, 1.0f,
            gates, nullptr, G4, BATCH, G4, D_H);
        lstm_cell_kernel<<<BATCH, blk, 0, stream>>>(
            gates, cbuf, enc + (size_t)t * BATCH * D_H, D_H, nullptr, nullptr, 0);
    }

    // ---- Phase 3: K/V projections (bf16 outputs: stay L2-resident) ----
    gemm_wmma_bf16<<<dim3(D_H / 128, TB / 128), blk, 0, stream>>>(
        enc, nullptr, D_H, attnw_b + (size_t)D_H * D_H, D_H, attn_in_b + D_H,
        nullptr, 0, 1.0f, nullptr, kbf, D_H, TB, D_H, D_H);
    gemm_wmma_bf16<<<dim3(D_H / 128, TB / 128), blk, 0, stream>>>(
        enc, nullptr, D_H, attnw_b + (size_t)2 * D_H * D_H, D_H, attn_in_b + 2 * D_H,
        nullptr, 0, 1.0f, nullptr, vbf, D_H, TB, D_H, D_H);

    // ---- Phase 4: decoder (serial 64 steps) ----
    init_decode<<<BATCH, blk, 0, stream>>>(xin, enc + (size_t)(T_STEPS - 1) * BATCH * D_H);
    const float scale = 0.0625f;  // 1/sqrt(256)
    for (int step = 0; step < MAXLEN; step++) {
        // q = scale * (out @ Wq^T + bq);  out = xin cols 0:256
        gemm_wmma_bf16<<<dim3(D_H / 128, BATCH / 128), blk, 0, stream>>>(
            xin, nullptr, 768, attnw_b, D_H, attn_in_b, nullptr, 0, scale,
            qbuf, nullptr, D_H, BATCH, D_H, D_H);
        // scores + softmax + weighted-V, fused per batch element
        attn_kernel<<<BATCH, blk, 0, stream>>>(qbuf, kbf, vbf, ctxb);
        // ctx2 = ctx @ attn_out_w^T + b  -> written into xin cols 256:512
        gemm_wmma_bf16<<<dim3(D_H / 128, BATCH / 128), blk, 0, stream>>>(
            ctxb, nullptr, D_H, attow_b, D_H, attn_out_b, nullptr, 0, 1.0f,
            nullptr, xin + 256, 768, BATCH, D_H, D_H);
        // gates = [out|ctx|h] @ [Wih|Whh]^T + (bih+bhh)
        gemm_wmma_bf16<<<dim3(G4 / 128, BATCH / 128), blk, 0, stream>>>(
            xin, nullptr, 768, decw_b, 768, decb, nullptr, 0, 1.0f,
            gates, nullptr, G4, BATCH, G4, 768);
        // cell: h -> xin cols 512:768, relu(h) -> outf + xin cols 0:256
        lstm_cell_kernel<<<BATCH, blk, 0, stream>>>(
            gates, cbuf, xin + 512, 768, outf, xin, 768);
        // logits
        logits_kernel<<<dim3((BATCH * VOCAB + 255) / 256), blk, 0, stream>>>(
            outf, lin_w, lin_b, logits_out + (size_t)step * BATCH * VOCAB);
    }
}
